// Encoder_31593779429711
// MI455X (gfx1250) — matrix operand order repaired and checked by reading, not compiled, not run on hardware
//
#include <hip/hip_runtime.h>
#include <cstdint>
#include <cstddef>

// Bidirectional LSTM encoder for MI455X (gfx1250, wave32, WMMA), v3.
//   T=100, B=2048, IN=128, H=100, OUT=20
// v3 change vs v2: LSTM activations use the native CDNA5 V_TANH_F32 trans op
// (guarded by __has_builtin); sigmoid computed as 0.5*tanh(0.5x)+0.5 so every
// gate activation is one TRANS + 2 VALU, co-executable with the WMMA pipe.
//
// Pipeline (all on `stream`):
//   1) xpack_kernel : x (f32) -> bf16 A-operand tiles        (52.4 MB in ws)
//   2) fcpack_kernel: fc_w slices -> bf16 B-operand tiles    ( 1.6 MB in ws)
//   3) bilstm_fused_kernel: 256 WGs (128 batch-tiles x 2 dirs), 4 waves each;
//      weights packed once to LDS (bf16 B-layout, gate-major cols, H 100->112),
//      c in registers, h double-buffered in LDS, FC fused on wave 3.
//   4) combine_kernel: out = fwd_partial + bwd_partial + fc_b.

#define T_STEPS 100
#define BATCH   2048
#define NBT     (BATCH/16)           // 128 batch tiles
#define IN_SZ   128
#define HID     100
#define OUT_SZ  20
#define HPAD    112                  // padded hidden per gate (7 tiles of 16)
#define KPAD    128                  // padded K for recurrent / FC A operand
#define NJT     7
#define NTILES  28                   // 4 gates * NJT
#define FCW_STRIDE (2*HID*T_STEPS)   // 20000

// d_ws layout (bytes)
#define PART_BYTES   ((size_t)2*BATCH*OUT_SZ*4)                   //   327,680
#define XPACK_U16    ((size_t)T_STEPS*NBT*4*32*16)                // 26,214,400
#define XPACK_BYTES  (XPACK_U16*2)                                // 52,428,800
#define FCPACK_U16   ((size_t)T_STEPS*2*8*32*16)                  //    819,200
#define FCPACK_BYTES (FCPACK_U16*2)                               //  1,638,400

typedef __attribute__((ext_vector_type(16))) __bf16 v16bf;
typedef __attribute__((ext_vector_type(8)))  float  v8f;
typedef __attribute__((ext_vector_type(8)))  int    v8i;

static __device__ __forceinline__ unsigned short f2bfu(float f) {
  union { float f; unsigned u; } v; v.f = f;
  unsigned r = v.u + 0x7FFFu + ((v.u >> 16) & 1u);   // round-to-nearest-even
  return (unsigned short)(r >> 16);
}

static __device__ __forceinline__ v16bf pack16(const float* s) {
  v8i w;
#pragma unroll
  for (int p = 0; p < 8; ++p)
    w[p] = (int)(((unsigned)f2bfu(s[2*p])) | (((unsigned)f2bfu(s[2*p+1])) << 16));
  return __builtin_bit_cast(v16bf, w);
}

// Native CDNA5 transcendental tanh when available (V_TANH_F32), else ocml.
static __device__ __forceinline__ float fast_tanh(float x) {
#if __has_builtin(__builtin_amdgcn_tanhf)
  return __builtin_amdgcn_tanhf(x);
#else
  return tanhf(x);
#endif
}
// sigmoid(x) == 0.5*tanh(x/2) + 0.5 (exact identity): 1 TRANS + 2 VALU.
static __device__ __forceinline__ float fast_sigmoid(float x) {
  return __builtin_fmaf(0.5f, fast_tanh(0.5f * x), 0.5f);
}

// 16 bf16 stored contiguously (32B/lane slot) -> v16bf. Works for LDS & global.
static __device__ __forceinline__ v16bf ld16_cont(const unsigned short* p) {
  uint4 a = *(const uint4*)(p);
  uint4 b = *(const uint4*)(p + 8);
  v8i w; w[0]=a.x; w[1]=a.y; w[2]=a.z; w[3]=a.w; w[4]=b.x; w[5]=b.y; w[6]=b.z; w[7]=b.w;
  return __builtin_bit_cast(v16bf, w);
}

// A-operand gather from row-major bf16: K..K+7 at p, K+16..K+23 at p+16.
static __device__ __forceinline__ v16bf ld16_gap(const unsigned short* p) {
  uint4 a = *(const uint4*)(p);
  uint4 b = *(const uint4*)(p + 16);
  v8i w; w[0]=a.x; w[1]=a.y; w[2]=a.z; w[3]=a.w; w[4]=b.x; w[5]=b.y; w[6]=b.z; w[7]=b.w;
  return __builtin_bit_cast(v16bf, w);
}

static __device__ __forceinline__ v8f wmma_bf16(v16bf a, v16bf b, v8f c) {
  return __builtin_amdgcn_wmma_f32_16x16x32_bf16(false, a, false, b, (short)0, c,
                                                 false, false);
}

// ---------------------------------------------------------------------------
// Pre-pass 1: x (T,B,IN) f32 -> bf16 A-operand tiles.
// Slot s = ((t*NBT + bt)*4 + kb)*32 + lane, 16 bf16 per slot.
// A layout (16-bit 16x32): lane<16: e0..7 -> K k0..k0+7, e8..15 -> K k0+16..23,
//                          lane>=16: k0 += 8.
__global__ __launch_bounds__(256)
void xpack_kernel(const float* __restrict__ x, unsigned short* __restrict__ xpack)
{
  const size_t s = (size_t)blockIdx.x * blockDim.x + threadIdx.x;
  if (s >= (size_t)T_STEPS*NBT*4*32) return;
  const int lane = (int)(s & 31);
  const int kb   = (int)((s >> 5) & 3);
  const int bt   = (int)((s >> 7) & (NBT - 1));
  const int t    = (int)(s >> 14);
  const int row  = lane & 15;
  const int khi  = lane >> 4;
  const int k0   = kb*32 + khi*8;
  const float* xr = x + ((size_t)t*BATCH + bt*16 + row)*IN_SZ + k0;
  float buf[16];
  const float4 f0 = *(const float4*)(xr);
  const float4 f1 = *(const float4*)(xr + 4);
  const float4 f2 = *(const float4*)(xr + 16);
  const float4 f3 = *(const float4*)(xr + 20);
  buf[0]=f0.x;  buf[1]=f0.y;  buf[2]=f0.z;  buf[3]=f0.w;
  buf[4]=f1.x;  buf[5]=f1.y;  buf[6]=f1.z;  buf[7]=f1.w;
  buf[8]=f2.x;  buf[9]=f2.y;  buf[10]=f2.z; buf[11]=f2.w;
  buf[12]=f3.x; buf[13]=f3.y; buf[14]=f3.z; buf[15]=f3.w;
  const v16bf v = pack16(buf);
  const v8i w = __builtin_bit_cast(v8i, v);
  uint4* dst = (uint4*)(xpack + s*16);
  dst[0] = make_uint4(w[0], w[1], w[2], w[3]);
  dst[1] = make_uint4(w[4], w[5], w[6], w[7]);
}

// ---------------------------------------------------------------------------
// Pre-pass 2: fc_w (OUT, 2*H*T) f32 -> bf16 B-operand tiles per (t,dir).
// Slot s = (((t*2 + dir)*8) + nt*4 + kb)*32 + lane, 16 bf16 per slot.
// B layout (16-bit 32x16): lane<16 holds col n, K=kb*32+e (e<8), kb*32+e+8 (e>=8); lane>=16: K += 8.
__global__ __launch_bounds__(256)
void fcpack_kernel(const float* __restrict__ fcw, unsigned short* __restrict__ fcpack)
{
  const size_t s = (size_t)blockIdx.x * blockDim.x + threadIdx.x;
  if (s >= (size_t)T_STEPS*2*8*32) return;
  const int lane = (int)(s & 31);
  const int kb   = (int)((s >> 5) & 3);
  const int nt   = (int)((s >> 7) & 1);
  const int dir  = (int)((s >> 8) & 1);
  const int t    = (int)(s >> 9);
  const int nl   = lane & 15;
  const int khi  = lane >> 4;
  const int o    = nt*16 + nl;
  const float* fsl = fcw + t*(2*HID) + dir*HID;
  float buf[16];
#pragma unroll
  for (int e = 0; e < 16; ++e) {
    const int j = kb*32 + khi*8 + (e < 8 ? e : e + 8);
    buf[e] = (o < OUT_SZ && j < HID) ? fsl[(size_t)o*FCW_STRIDE + j] : 0.f;
  }
  const v16bf v = pack16(buf);
  const v8i w = __builtin_bit_cast(v8i, v);
  uint4* dst = (uint4*)(fcpack + s*16);
  dst[0] = make_uint4(w[0], w[1], w[2], w[3]);
  dst[1] = make_uint4(w[4], w[5], w[6], w[7]);
}

// ---------------------------------------------------------------------------
__global__ __launch_bounds__(128, 1)
void bilstm_fused_kernel(const unsigned short* __restrict__ xpack,
                         const unsigned short* __restrict__ fcpack,
                         const float* __restrict__ Wih_f, const float* __restrict__ Whh_f,
                         const float* __restrict__ bih_f, const float* __restrict__ bhh_f,
                         const float* __restrict__ Wih_b, const float* __restrict__ Whh_b,
                         const float* __restrict__ bih_b, const float* __restrict__ bhh_b,
                         float* __restrict__ ws_partial)
{
  extern __shared__ unsigned char smem[];
  unsigned short* wihL  = (unsigned short*)smem;                // 28*4*32*16 u16
  unsigned short* whhL  = wihL + NTILES*4*32*16;                // same size
  float*          biasL = (float*)(whhL + NTILES*4*32*16);      // 4*HPAD f32
  unsigned short* hbuf  = (unsigned short*)(biasL + 4*HPAD);    // 2 * 16*KPAD u16

  const int dir = blockIdx.x & 1;
  const int bt  = blockIdx.x >> 1;
  const int bbase = bt * 16;
  const float* Wih = dir ? Wih_b : Wih_f;
  const float* Whh = dir ? Whh_b : Whh_f;
  const float* bih = dir ? bih_b : bih_f;
  const float* bhh = dir ? bhh_b : bhh_f;

  const int tid = threadIdx.x;

  // One-time LDS weight packing (f32 -> bf16, B-operand layout; cols permuted
  // to gate*HPAD + j so each 16-wide N-tile is a single gate).
  for (int idx = tid; idx < NTILES*4*32*16; idx += blockDim.x) {
    const int e    = idx & 15;
    const int lane = (idx >> 4) & 31;
    const int kb   = (idx >> 9) & 3;
    const int tile = idx >> 11;
    const int colp = tile*16 + (lane & 15);
    const int gate = colp / HPAD;
    const int j    = colp % HPAD;
    const int K    = kb*32 + (lane >> 4)*8 + (e < 8 ? e : e + 8);
    const float vi = (j < HID)            ? Wih[(gate*HID + j)*IN_SZ + K] : 0.f;
    const float vh = (j < HID && K < HID) ? Whh[(gate*HID + j)*HID  + K] : 0.f;
    wihL[idx] = f2bfu(vi);
    whhL[idx] = f2bfu(vh);
  }
  for (int idx = tid; idx < 4*HPAD; idx += blockDim.x) {
    const int gate = idx / HPAD, j = idx % HPAD;
    biasL[idx] = (j < HID) ? (bih[gate*HID + j] + bhh[gate*HID + j]) : 0.f;
  }
  for (int idx = tid; idx < 2*16*KPAD; idx += blockDim.x) hbuf[idx] = 0;
  __syncthreads();

  const int wv   = tid >> 5;
  const int lane = tid & 31;
  const int nl   = lane & 15;
  const int khi  = lane >> 4;

  float c01[2][8];
#pragma unroll
  for (int a = 0; a < 2; ++a)
#pragma unroll
    for (int r = 0; r < 8; ++r) c01[a][r] = 0.f;
  v8f fca0 = {}, fca1 = {};

  int cur = 0;
  for (int s = 0; s < T_STEPS; ++s) {
    const int t = dir ? (T_STEPS - 1 - s) : s;
    unsigned short* hc = hbuf + cur * 16 * KPAD;
    unsigned short* hn = hbuf + (1 - cur) * 16 * KPAD;

    // A operands: pure b128 loads (no conversion VALU on the serial path).
    v16bf xa[4], ha[4];
    const unsigned short* xp = xpack + ((size_t)(t*NBT + bt)*4)*32*16;
#pragma unroll
    for (int kb = 0; kb < 4; ++kb) {
      xa[kb] = ld16_cont(xp + (kb*32 + lane)*16);
      ha[kb] = ld16_gap(hc + nl * KPAD + kb*32 + khi*8);
    }

    // Wave 3: fold previous step's h into FC (pipelined one step behind).
    if (wv == 3 && s > 0) {
      const int tp = dir ? (T_STEPS - s) : (s - 1);
      const unsigned short* fp = fcpack + ((size_t)(tp*2 + dir)*8)*32*16;
#pragma unroll
      for (int nt = 0; nt < 2; ++nt) {
#pragma unroll
        for (int kb = 0; kb < 4; ++kb) {
          const v16bf bv = ld16_cont(fp + ((nt*4 + kb)*32 + lane)*16);
          if (nt == 0) fca0 = wmma_bf16(ha[kb], bv, fca0);
          else         fca1 = wmma_bf16(ha[kb], bv, fca1);
        }
      }
    }

    // Gate GEMMs + LSTM state update (waves split the 7 hidden tiles).
#pragma unroll
    for (int jj = 0; jj < 2; ++jj) {
      if (wv == 3 && jj == 1) continue;
      const int jt = (wv < 3) ? (2*wv + jj) : 6;
      v8f acc[4];
#pragma unroll
      for (int g = 0; g < 4; ++g) {
        const float bvv = biasL[g*HPAD + jt*16 + nl];
#pragma unroll
        for (int r = 0; r < 8; ++r) acc[g][r] = bvv;
      }
#pragma unroll
      for (int g = 0; g < 4; ++g) {
        const int tile = g*NJT + jt;
#pragma unroll
        for (int kb = 0; kb < 4; ++kb)
          acc[g] = wmma_bf16(xa[kb],
                             ld16_cont(wihL + ((tile*4 + kb)*32 + lane)*16), acc[g]);
#pragma unroll
        for (int kb = 0; kb < 4; ++kb)
          acc[g] = wmma_bf16(ha[kb],
                             ld16_cont(whhL + ((tile*4 + kb)*32 + lane)*16), acc[g]);
      }
      // Activations in D layout; native V_TANH_F32 trans ops.
#pragma unroll
      for (int r = 0; r < 8; ++r) {
        const float ig = fast_sigmoid(acc[0][r]);
        const float fg = fast_sigmoid(acc[1][r]);
        const float gg = fast_tanh(acc[2][r]);
        const float og = fast_sigmoid(acc[3][r]);
        const float cc = __builtin_fmaf(fg, c01[jj][r], ig * gg);
        c01[jj][r] = cc;
        const float hh = og * fast_tanh(cc);
        hn[(r + 8*khi)*KPAD + jt*16 + nl] = f2bfu(hh);
      }
    }
    __syncthreads();
    cur ^= 1;
  }

  // Final FC step (last timestep's h) + write partial outputs.
  if (wv == 3) {
    const unsigned short* hc = hbuf + cur * 16 * KPAD;
    v16bf hl[4];
#pragma unroll
    for (int kb = 0; kb < 4; ++kb)
      hl[kb] = ld16_gap(hc + nl * KPAD + kb*32 + khi*8);
    const int tp = dir ? 0 : (T_STEPS - 1);
    const unsigned short* fp = fcpack + ((size_t)(tp*2 + dir)*8)*32*16;
#pragma unroll
    for (int nt = 0; nt < 2; ++nt) {
#pragma unroll
      for (int kb = 0; kb < 4; ++kb) {
        const v16bf bv = ld16_cont(fp + ((nt*4 + kb)*32 + lane)*16);
        if (nt == 0) fca0 = wmma_bf16(hl[kb], bv, fca0);
        else         fca1 = wmma_bf16(hl[kb], bv, fca1);
      }
    }
    float* wsd = ws_partial + (size_t)dir * BATCH * OUT_SZ;
#pragma unroll
    for (int r = 0; r < 8; ++r) {
      const int b = bbase + r + 8*khi;
      wsd[b*OUT_SZ + nl] = fca0[r];
      if (16 + nl < OUT_SZ) wsd[b*OUT_SZ + 16 + nl] = fca1[r];
    }
  }
}

__global__ void bilstm_combine_kernel(const float* __restrict__ ws_partial,
                                      const float* __restrict__ fcb,
                                      float* __restrict__ out)
{
  const int i = blockIdx.x * blockDim.x + threadIdx.x;
  if (i < BATCH * OUT_SZ)
    out[i] = ws_partial[i] + ws_partial[BATCH*OUT_SZ + i] + fcb[i % OUT_SZ];
}

extern "C" void kernel_launch(void* const* d_in, const int* in_sizes, int n_in,
                              void* d_out, int out_size, void* d_ws, size_t ws_size,
                              hipStream_t stream) {
  (void)in_sizes; (void)n_in; (void)out_size; (void)ws_size;
  const float* x     = (const float*)d_in[0];
  const float* Wih_f = (const float*)d_in[1];
  const float* Whh_f = (const float*)d_in[2];
  const float* bih_f = (const float*)d_in[3];
  const float* bhh_f = (const float*)d_in[4];
  const float* Wih_b = (const float*)d_in[5];
  const float* Whh_b = (const float*)d_in[6];
  const float* bih_b = (const float*)d_in[7];
  const float* bhh_b = (const float*)d_in[8];
  const float* fcw   = (const float*)d_in[9];
  const float* fcb   = (const float*)d_in[10];
  float* out = (float*)d_out;

  float*          partials = (float*)d_ws;
  unsigned short* xpack    = (unsigned short*)((char*)d_ws + PART_BYTES);
  unsigned short* fcpack   = (unsigned short*)((char*)d_ws + PART_BYTES + XPACK_BYTES);

  // 1) operand pre-packing (fully parallel; off the recurrent critical path)
  {
    const size_t n = (size_t)T_STEPS*NBT*4*32;           // 1,638,400 slots
    xpack_kernel<<<dim3((unsigned)((n + 255)/256)), dim3(256), 0, stream>>>(x, xpack);
  }
  {
    const size_t n = (size_t)T_STEPS*2*8*32;             // 51,200 slots
    fcpack_kernel<<<dim3((unsigned)((n + 255)/256)), dim3(256), 0, stream>>>(fcw, fcpack);
  }

  // 2) fused recurrent kernel
  constexpr size_t LDS_BYTES =
      (size_t)NTILES*4*32*16*2 * 2   // Wih + Whh bf16 packs (229,376 B)
      + 4*HPAD*4                     // combined bias        (  1,792 B)
      + 2*16*KPAD*2;                 // h double buffer      (  8,192 B)
                                     // total 239,360 B < 320 KB/WGP
  bilstm_fused_kernel<<<dim3(NBT*2), dim3(128), LDS_BYTES, stream>>>(
      xpack, fcpack, Wih_f, Whh_f, bih_f, bhh_f, Wih_b, Whh_b, bih_b, bhh_b,
      partials);

  // 3) combine partials + bias
  bilstm_combine_kernel<<<dim3((BATCH*OUT_SZ + 255)/256), dim3(256), 0, stream>>>(
      partials, fcb, out);
}
